// RNNModel_30142080484084
// MI455X (gfx1250) — matrix operand order repaired and checked by reading, not compiled, not run on hardware
//
#include <hip/hip_runtime.h>

typedef _Float16 __attribute__((ext_vector_type(16))) v16h;
typedef float    __attribute__((ext_vector_type(8)))  v8f;
typedef float    __attribute__((ext_vector_type(4)))  v4f;
typedef int      __attribute__((ext_vector_type(8)))  v8i;
typedef int      __attribute__((ext_vector_type(4)))  v4i;
typedef unsigned __attribute__((ext_vector_type(2)))  v2u;

#define B_TOT 8192
#define T_TOT 512
#define IN    8
#define HID   16
#define WAVES 4
#define CHUNK 16

// gfx1250 has a hardware tanh transcendental (V_TANH_F32).
__device__ __forceinline__ float tanh_hw(float x) {
  float y;
  asm("v_tanh_f32 %0, %1" : "=v"(y) : "v"(x));
  return y;
}

// One wave owns 16 batch rows. State kept transposed: G = H^T (hidden x batch).
// Per step:  G_new = tanh( [W_hh | W_ih | 0] (16x32)  @  [G ; X_t^T ; 0] (32x16) + bias )
// -> exactly one v_wmma_f32_16x16x32_f16 per timestep, A and C loop-invariant.
__global__ __launch_bounds__(WAVES * 32) void rnn_wmma_kernel(
    const float* __restrict__ x,   const float* __restrict__ Wih,
    const float* __restrict__ Whh, const float* __restrict__ bih,
    const float* __restrict__ bhh, const float* __restrict__ fcw,
    const float* __restrict__ fcb, float* __restrict__ out)
{
  __shared__ _Float16 xstage[WAVES][16][CHUNK][IN];   // 4 KB per wave, f16

  const int  tid  = threadIdx.x;
  const int  wv   = tid >> 5;
  const int  lane = tid & 31;
  const int  l16  = lane & 15;
  const bool lo   = lane < 16;
  const int  b0   = (blockIdx.x * WAVES + wv) * 16;

  // ---- A fragment (constant): 16x32 f16 = [W_hh | W_ih | 0]
  // A k order used here (contiguous, same as B): lanes 0-15 -> VGPR0-7:K0..15
  //                  lanes16-31 -> VGPR0-7:K16..31
  v16h a;
  {
    const float* whrow = Whh + l16 * HID;
    const float* wirow = Wih + l16 * IN;
#pragma unroll
    for (int j = 0; j < 8; ++j) a[j] = lo ? (_Float16)whrow[j] : (_Float16)wirow[j];
#pragma unroll
    for (int j = 0; j < 8; ++j) a[8 + j] = lo ? (_Float16)whrow[8 + j] : (_Float16)0.0f;
  }

  // ---- C fragment (constant): bias per output row m (m = VGPR r + 8*hiHalf)
  const int moff = lo ? 0 : 8;
  v8f c;
#pragma unroll
  for (int r = 0; r < 8; ++r) c[r] = bih[moff + r] + bhh[moff + r];

  const int xswap = (lane ^ 16) << 2;   // ds_bpermute byte index: swap lane halves

  v4i  pk = {0, 0, 0, 0};               // own half of G columns (f16 pairs along m)
  v4i  sw = {0, 0, 0, 0};               // partner half (from the other 16 lanes)
  float hcur[8];
#pragma unroll
  for (int r = 0; r < 8; ++r) hcur[r] = 0.0f;

  const v4i* xrd = (const v4i*)&xstage[wv][l16][0][0];   // 1 timestep = 8 f16 = b128

  for (int t0 = 0; t0 < T_TOT; t0 += CHUNK) {
    // ---- stage CHUNK timesteps of x for the 16 rows: coalesced 512B per row,
    //      converted f32->f16 in flight; prefetch the next chunk.
#pragma unroll 4
    for (int i = 0; i < 16; ++i) {
      const v4f* src = (const v4f*)(x + ((size_t)(b0 + i) * T_TOT + t0) * IN);
      v4f v = src[lane];
      if (t0 + CHUNK < T_TOT) __builtin_prefetch(src + 32 + lane, 0, 0);
      v2u p;
      p.x = __builtin_bit_cast(unsigned, __builtin_amdgcn_cvt_pkrtz(v[0], v[1]));
      p.y = __builtin_bit_cast(unsigned, __builtin_amdgcn_cvt_pkrtz(v[2], v[3]));
      *((v2u*)&xstage[wv][i][0][0] + lane) = p;
    }

    // ---- serial recurrence over the staged chunk: 1 WMMA per step
#pragma unroll
    for (int t = 0; t < CHUNK; ++t) {
      v4i xf = xrd[t];                  // ds_load_b128 (lanes<16 value unused)
      // B layout: lanes 0-15 hold K0..15 (G column n); lanes 16-31 hold
      // K16..31 = [X_t^T column n ; zeros].
      v8i bi;
      bi[0] = lo ? pk[0] : xf[0];
      bi[1] = lo ? pk[1] : xf[1];
      bi[2] = lo ? pk[2] : xf[2];
      bi[3] = lo ? pk[3] : xf[3];
      bi[4] = lo ? sw[0] : 0;
      bi[5] = lo ? sw[1] : 0;
      bi[6] = lo ? sw[2] : 0;
      bi[7] = lo ? sw[3] : 0;
      v16h b = __builtin_bit_cast(v16h, bi);

      v8f d = __builtin_amdgcn_wmma_f32_16x16x32_f16(
          false, a, false, b, (short)0, c, false, false);

#pragma unroll
      for (int r = 0; r < 8; ++r) hcur[r] = tanh_hw(d[r]);   // independent TRANS chain

      pk[0] = __builtin_bit_cast(int, __builtin_amdgcn_cvt_pkrtz(hcur[0], hcur[1]));
      pk[1] = __builtin_bit_cast(int, __builtin_amdgcn_cvt_pkrtz(hcur[2], hcur[3]));
      pk[2] = __builtin_bit_cast(int, __builtin_amdgcn_cvt_pkrtz(hcur[4], hcur[5]));
      pk[3] = __builtin_bit_cast(int, __builtin_amdgcn_cvt_pkrtz(hcur[6], hcur[7]));
      sw[0] = __builtin_amdgcn_ds_bpermute(xswap, pk[0]);
      sw[1] = __builtin_amdgcn_ds_bpermute(xswap, pk[1]);
      sw[2] = __builtin_amdgcn_ds_bpermute(xswap, pk[2]);
      sw[3] = __builtin_amdgcn_ds_bpermute(xswap, pk[3]);
    }
  }

  // ---- fc head: out[n] = sum_m fcw[m] * h[n][m] + fcb
  float partial = 0.0f;
#pragma unroll
  for (int r = 0; r < 8; ++r) partial += fcw[moff + r] * hcur[r];
  float other = __builtin_bit_cast(
      float, __builtin_amdgcn_ds_bpermute(xswap, __builtin_bit_cast(int, partial)));
  if (lo) out[b0 + l16] = partial + other + fcb[0];
}

extern "C" void kernel_launch(void* const* d_in, const int* in_sizes, int n_in,
                              void* d_out, int out_size, void* d_ws, size_t ws_size,
                              hipStream_t stream) {
  (void)in_sizes; (void)n_in; (void)out_size; (void)d_ws; (void)ws_size;
  const float* x   = (const float*)d_in[0];
  const float* Wih = (const float*)d_in[1];
  const float* Whh = (const float*)d_in[2];
  const float* bih = (const float*)d_in[3];
  const float* bhh = (const float*)d_in[4];
  const float* fcw = (const float*)d_in[5];
  const float* fcb = (const float*)d_in[6];
  float* out = (float*)d_out;

  dim3 grid(B_TOT / (16 * WAVES));   // 128 blocks
  dim3 block(WAVES * 32);            // 4 waves -> 512 tiles total
  hipLaunchKernelGGL(rnn_wmma_kernel, grid, block, 0, stream,
                     x, Wih, Whh, bih, bhh, fcw, fcb, out);
}
